// RelationalCritic_56916906606807
// MI455X (gfx1250) — matrix operand order repaired and checked by reading, not compiled, not run on hardware
//
#include <hip/hip_runtime.h>
#include <hip/hip_bf16.h>

// ---------------- problem constants ----------------
#define C_NA   4
#define C_B    1024
#define C_NOBJ 128
#define C_F    16
#define C_H    64
#define C_R    8
#define C_A    13
#define C_E0   512
#define C_ETOT (C_B * C_E0)   // 524288 (edge_index second row offset)

typedef __attribute__((ext_vector_type(16))) __bf16 v16bf;
typedef __attribute__((ext_vector_type(8)))  __bf16 v8bf;
typedef __attribute__((ext_vector_type(4)))  __bf16 v4bf;
typedef __attribute__((ext_vector_type(2)))  __bf16 v2bf;
typedef __attribute__((ext_vector_type(8)))  float  v8f;
typedef __attribute__((ext_vector_type(4)))  float  v4f;
typedef unsigned int u32x4 __attribute__((ext_vector_type(4)));
typedef int          i32x4 __attribute__((ext_vector_type(4)));
typedef int          i32x8 __attribute__((ext_vector_type(8)));

#if defined(__has_builtin)
#  if __has_builtin(__builtin_amdgcn_tensor_load_to_lds)
#    define HAVE_TDM 1
#  endif
#  if __has_builtin(__builtin_amdgcn_cvt_pk_bf16_f32)
#    define HAVE_CVT_PK 1
#  endif
#  if __has_builtin(__builtin_amdgcn_perm)
#    define HAVE_PERM 1
#  endif
#endif

// packed float pair -> bf16 pair.
// Preferred: hw packed convert. Else: +0x8000 round-nearest(ties-away) on each
// float and a single v_perm_b32 to splice the two high halves.
__device__ __forceinline__ v2bf cvt2(float lo, float hi) {
#ifdef HAVE_CVT_PK
    auto t = __builtin_amdgcn_cvt_pk_bf16_f32(lo, hi);
    return __builtin_bit_cast(v2bf, t);
#else
    unsigned u0 = __builtin_bit_cast(unsigned, lo) + 0x8000u;
    unsigned u1 = __builtin_bit_cast(unsigned, hi) + 0x8000u;
#  ifdef HAVE_PERM
    // result bytes: [u1.b3, u1.b2, u0.b3, u0.b2]  (src0 = u1 -> codes 4..7)
    unsigned p = __builtin_amdgcn_perm(u1, u0, 0x07060302u);
#  else
    unsigned p = (u0 >> 16) | (u1 & 0xFFFF0000u);
#  endif
    return __builtin_bit_cast(v2bf, p);
#endif
}

__device__ __forceinline__ v16bf pack16(v2bf c0, v2bf c1, v2bf c2, v2bf c3,
                                        v2bf c4, v2bf c5, v2bf c6, v2bf c7) {
    v4bf q0 = __builtin_shufflevector(c0, c1, 0, 1, 2, 3);
    v4bf q1 = __builtin_shufflevector(c2, c3, 0, 1, 2, 3);
    v4bf q2 = __builtin_shufflevector(c4, c5, 0, 1, 2, 3);
    v4bf q3 = __builtin_shufflevector(c6, c7, 0, 1, 2, 3);
    v8bf h0 = __builtin_shufflevector(q0, q1, 0, 1, 2, 3, 4, 5, 6, 7);
    v8bf h1 = __builtin_shufflevector(q2, q3, 0, 1, 2, 3, 4, 5, 6, 7);
    return __builtin_shufflevector(h0, h1, 0, 1, 2, 3, 4, 5, 6, 7,
                                   8, 9, 10, 11, 12, 13, 14, 15);
}

__device__ __forceinline__ v8f wmma_bf16(v16bf a, v16bf b, v8f c) {
    // D(16x16,f32) = A(16x32,bf16) * B(32x16,bf16) + C
    return __builtin_amdgcn_wmma_f32_16x16x32_bf16(
        false, a, false, b, (short)0, c, false, false);
}

// ---- A fragment (16x32 bf16) from fp32 LDS, STRIDE = multiple of 4 floats ----
template <int STRIDE>
__device__ __forceinline__ v16bf load_a_f32v(const float* base, int mrow0, int kb, int lane) {
    int r = lane & 15, hf = lane >> 4;
    const float* p = base + (mrow0 + r) * STRIDE + kb + hf * 8;
    v4f x0 = *(const v4f*)(p);
    v4f x1 = *(const v4f*)(p + 4);
    v4f x2 = *(const v4f*)(p + 16);
    v4f x3 = *(const v4f*)(p + 20);
    return pack16(cvt2(x0[0], x0[1]), cvt2(x0[2], x0[3]),
                  cvt2(x1[0], x1[1]), cvt2(x1[2], x1[3]),
                  cvt2(x2[0], x2[1]), cvt2(x2[2], x2[3]),
                  cvt2(x3[0], x3[1]), cvt2(x3[2], x3[3]));
}

// ---- A fragment from bf16 LDS (16B-aligned stride) ----
template <int STRIDE>
__device__ __forceinline__ v16bf load_a_bf16v(const __bf16* base, int mrow0, int kb, int lane) {
    int r = lane & 15, hf = lane >> 4;
    const __bf16* p = base + (mrow0 + r) * STRIDE + kb + hf * 8;
    v8bf lo = *(const v8bf*)(p);
    v8bf hi = *(const v8bf*)(p + 16);
    return __builtin_shufflevector(lo, hi, 0, 1, 2, 3, 4, 5, 6, 7,
                                   8, 9, 10, 11, 12, 13, 14, 15);
}

// ---- A fragment for the root term: K 0..15 from Xs (stride 16 f32), K 16..31 = 0 ----
__device__ __forceinline__ v16bf load_a_x(const float* Xs, int mrow0, int lane) {
    int r = lane & 15, hf = lane >> 4;
    const float* p = Xs + (mrow0 + r) * 16 + hf * 8;
    v4f x0 = *(const v4f*)(p);
    v4f x1 = *(const v4f*)(p + 4);
    v2bf z = __builtin_bit_cast(v2bf, 0u);
    return pack16(cvt2(x0[0], x0[1]), cvt2(x0[2], x0[3]),
                  cvt2(x1[0], x1[1]), cvt2(x1[2], x1[3]), z, z, z, z);
}

// ---- B fragment from fragment-linear swizzled LDS: lane reads 16 contiguous bf16 ----
// layout [kbi][t][lane][elem]; element (row,col): row = kbi*32 + (e>>3)*16 + (lane>>4)*8 + (e&7),
// col = t*16 + (lane&15)
__device__ __forceinline__ v16bf load_b_frag(const __bf16* Wsw, int kbi, int t, int lane) {
    return *((const v16bf*)Wsw + (kbi * 4 + t) * 32 + lane);
}

// ---------------------------------------------------------------------------
// Kernel 0: per-edge mean weights  w_e = 1/cnt(dst,rel)  (base graph, batch-invariant)
// ---------------------------------------------------------------------------
__global__ __launch_bounds__(C_E0) void edge_weight_kernel(
    const int* __restrict__ edge_index, const int* __restrict__ edge_attr,
    float* __restrict__ wts) {
    __shared__ int cnt[C_NOBJ * C_R];
    int tid = threadIdx.x;
    for (int i = tid; i < C_NOBJ * C_R; i += C_E0) cnt[i] = 0;
    __syncthreads();
    int dst = edge_index[C_ETOT + tid];
    int rel = edge_attr[tid];
    atomicAdd(&cnt[dst * C_R + rel], 1);
    __syncthreads();
    wts[tid] = 1.0f / (float)cnt[dst * C_R + rel];
}

// ---------------------------------------------------------------------------
// Kernel 1: RGCN + relu + max-pool for one (agent, batch) pair per workgroup.
//   agg = Y @ Wcat + X @ root ; Y[n, r*16+f] = sum_{e:dst=n,rel=r} w_e * X[src_e, f]
//   pooled[a,b,:] = max_n relu(agg + bias)
// ---------------------------------------------------------------------------
#define YS 68  // fp32 row stride for Y: 272B (16B aligned, conflict-free A loads)

__global__ __launch_bounds__(256) void rgcn_pool_kernel(
    const float* __restrict__ unary, const int* __restrict__ edge_index,
    const int* __restrict__ edge_attr, const float* __restrict__ Wg,
    const float* __restrict__ rootg, const float* __restrict__ biasg,
    const float* __restrict__ wts, float* __restrict__ pooled) {
    __shared__ __align__(32) float  Ys[C_NOBJ * YS];     // 34,816 B (scatter, 4 rel/half)
    __shared__ __align__(16) float  Xs[C_NOBJ * C_F];    //  8,192 B (node features)
    __shared__ __align__(32) __bf16 Ws[5 * 4 * 32 * 16]; // 20,480 B frag-swizzled Wcat+root

    const int tid  = threadIdx.x;
    const int lane = tid & 31;
    const int wave = tid >> 5;
    const int blk  = blockIdx.x;        // a * B + b
    const int a    = blk >> 10;
    const int b    = blk & (C_B - 1);

    const float* xg = unary + (size_t)(a * C_B + b) * (C_NOBJ * C_F);

#ifdef HAVE_TDM
    // --- Tensor Data Mover: async DMA the 8KB X tile (2048 fp32) into LDS ---
    if (wave == 0) {
        unsigned long long ga = (unsigned long long)(uintptr_t)xg;
        unsigned ldsOff =
            (unsigned)(unsigned long long)(__attribute__((address_space(3))) float*)Xs;
        // D# group0: count=1, lds_addr, 57b global_addr, type=2
        u32x4 g0 = {1u, ldsOff, (unsigned)ga,
                    (unsigned)(ga >> 32) | 0x80000000u};
        // D# group1: data_size=4B(code 2); tensor_dim0=2048, tensor_dim1=1;
        //            tile_dim0=2048, tile_dim1=1; tensor_dim0_stride=2048
        i32x8 g1 = {(int)(2u << 16),     // data_size
                    (int)(2048u << 16),  // tensor_dim0[15:0] << 16
                    (int)(1u << 16),     // tensor_dim0 hi=0 | tensor_dim1[15:0]<<16
                    (int)(2048u << 16),  // tensor_dim1 hi=0 | tile_dim0<<16
                    1,                   // tile_dim1=1, tile_dim2=0
                    2048,                // tensor_dim0_stride lo32
                    0, 0};
        i32x4 gz = {0, 0, 0, 0};
#  if __has_include(<hip/amd_detail/amd_gfx1250_TDM.h>)
        i32x8 gz8 = {0, 0, 0, 0, 0, 0, 0, 0};
        __builtin_amdgcn_tensor_load_to_lds(g0, g1, gz, gz, gz8, 0);
#  else
        __builtin_amdgcn_tensor_load_to_lds(g0, g1, gz, gz, 0);
#  endif
    }
#else
    for (int i = tid; i < C_NOBJ * C_F; i += 256) Xs[i] = xg[i];
#endif

    // Wcat (rows 0-127 = W[r][f], 128-143 = root, 144-159 = 0), fragment-linear,
    // converted two elements (consecutive rows e, e+1) per iteration.
    for (int ii = tid; ii < 5 * 4 * 32 * 16 / 2; ii += 256) {
        int i = ii * 2;
        int e = i & 15, l = (i >> 4) & 31, t = (i >> 9) & 3, kbi = i >> 11;
        int row = kbi * 32 + (e >> 3) * 16 + (l >> 4) * 8 + (e & 7);
        int col = t * 16 + (l & 15);
        float v0, v1;
        if (row < 128)       v0 = Wg[row * C_H + col];
        else if (row < 144)  v0 = rootg[(row - 128) * C_H + col];
        else                 v0 = 0.0f;
        int row1 = row + 1;
        if (row1 < 128)      v1 = Wg[row1 * C_H + col];
        else if (row1 < 144) v1 = rootg[(row1 - 128) * C_H + col];
        else                 v1 = 0.0f;
        *(v2bf*)&Ws[i] = cvt2(v0, v1);
    }
    for (int i = tid; i < C_NOBJ * YS; i += 256) Ys[i] = 0.0f;
#ifdef HAVE_TDM
    if (wave == 0) __builtin_amdgcn_s_wait_tensorcnt(0);
#endif
    __syncthreads();

    const v8f vzero = {0.f, 0.f, 0.f, 0.f, 0.f, 0.f, 0.f, 0.f};
    v8f acc[4] = {vzero, vzero, vzero, vzero};
    const int mrow0 = wave * 16;

    for (int hfrel = 0; hfrel < 2; ++hfrel) {
        // sparse scatter: edges with rel in [4*hfrel, 4*hfrel+4)
        for (int e = tid; e < C_E0; e += 256) {
            int rel = edge_attr[e];
            if ((rel >> 2) == hfrel) {
                int   src = edge_index[e];
                int   dst = edge_index[C_ETOT + e];
                float w   = wts[e];
                int   cb  = (rel & 3) * 16;
#pragma unroll
                for (int f = 0; f < 16; ++f)
                    atomicAdd(&Ys[dst * YS + cb + f], w * Xs[src * 16 + f]);
            }
        }
        __syncthreads();
        // dense: acc += Y[:,0:64] @ Wcat[64*hfrel .. 64*hfrel+64, :]
#pragma unroll
        for (int kb = 0; kb < 64; kb += 32) {
            v16bf afrag = load_a_f32v<YS>(Ys, mrow0, kb, lane);
            int   kbi   = hfrel * 2 + (kb >> 5);
#pragma unroll
            for (int t = 0; t < 4; ++t)
                acc[t] = wmma_bf16(afrag, load_b_frag(Ws, kbi, t, lane), acc[t]);
        }
        __syncthreads();
        if (hfrel == 0) {
            for (int i = tid; i < C_NOBJ * YS; i += 256) Ys[i] = 0.0f;
            __syncthreads();
        }
    }
    // root term: acc += X (K=16 zero-padded) @ root (swizzle block kbi=4)
    {
        v16bf afrag = load_a_x(Xs, mrow0, lane);
#pragma unroll
        for (int t = 0; t < 4; ++t)
            acc[t] = wmma_bf16(afrag, load_b_frag(Ws, 4, t, lane), acc[t]);
    }
    // bias + relu + row-max; reduction buffer aliases retired Ys space
    float* smax = Ys;  // 8*2*64 = 1024 floats
    const int c15 = lane & 15, hf = lane >> 4;
#pragma unroll
    for (int t = 0; t < 4; ++t) {
        float bb = biasg[t * 16 + c15];
        float m  = -1e30f;
#pragma unroll
        for (int v = 0; v < 8; ++v) {
            float x = acc[t][v] + bb;
            x = x > 0.f ? x : 0.f;
            m = x > m ? x : m;
        }
        smax[wave * 128 + hf * 64 + t * 16 + c15] = m;
    }
    __syncthreads();
    if (tid < C_H) {
        float m = -1e30f;
#pragma unroll
        for (int w = 0; w < 16; ++w) {
            float x = smax[w * 64 + tid];
            m = x > m ? x : m;
        }
        pooled[(size_t)(a * C_B + b) * C_H + tid] = m;
    }
}

// ---------------------------------------------------------------------------
// Kernel 2: per-agent MLP head.  One workgroup = (agent, 128 batch rows).
//   h1 = leaky_relu(cin @ fc1_w[a]^T + fc1_b[a]);  q = h1 . fc2_w[a][argmax(actions[a])]
// ---------------------------------------------------------------------------
#define CS 136  // bf16 row stride for cin: 272B (16B aligned)

__global__ __launch_bounds__(256) void mlp_kernel(
    const float* __restrict__ actions, const float* __restrict__ fc1w,
    const float* __restrict__ fc1b, const float* __restrict__ fc2w,
    const float* __restrict__ fc2b, const float* __restrict__ pooled,
    float* __restrict__ out) {
    __shared__ __align__(32) char   smemc[128 * CS * 2];   // cin(bf16) -> reused as h1(f32)
    __shared__ __align__(32) __bf16 w1t[4 * 4 * 32 * 16];  // fc1_w[a]^T frag-swizzled, K->128

    __bf16* cin = (__bf16*)smemc;
    float*  h1  = (float*)smemc;  // 128*64 f32 = 32,768 B <= 34,816 B

    const int tid  = threadIdx.x;
    const int lane = tid & 31;
    const int wave = tid >> 5;
    const int a    = blockIdx.y;
    const int b0   = blockIdx.x * 128;

    // cin rows: [pooled(64) | other agents' actions(39) | zero pad], two cols per iter
    for (int ii = tid; ii < 128 * CS / 2; ii += 256) {
        int i   = ii * 2;
        int row = i / CS, col = i - row * CS;
        int b   = b0 + row;
        float v[2];
#pragma unroll
        for (int s = 0; s < 2; ++s) {
            int c = col + s;
            float x = 0.0f;
            if (c < 64) {
                x = pooled[((size_t)a * C_B + b) * C_H + c];
            } else if (c < 64 + C_A * (C_NA - 1)) {
                int k  = c - 64;
                int jj = k / C_A, kk = k - jj * C_A;
                int j  = jj + (jj >= a ? 1 : 0);
                x = actions[((size_t)j * C_B + b) * C_A + kk];
            }
            v[s] = x;
        }
        *(v2bf*)&cin[i] = cvt2(v[0], v[1]);
    }
    // fc1_w[a]^T in fragment-linear order (K padded 103 -> 128), two rows per iter
    for (int ii = tid; ii < 4 * 4 * 32 * 16 / 2; ii += 256) {
        int i = ii * 2;
        int e = i & 15, l = (i >> 4) & 31, t = (i >> 9) & 3, kbi = i >> 11;
        int k = kbi * 32 + (e >> 3) * 16 + (l >> 4) * 8 + (e & 7);
        int n = t * 16 + (l & 15);
        float v0 = (k < 103)     ? fc1w[((size_t)a * C_H + n) * 103 + k]     : 0.0f;
        float v1 = (k + 1 < 103) ? fc1w[((size_t)a * C_H + n) * 103 + k + 1] : 0.0f;
        *(v2bf*)&w1t[i] = cvt2(v0, v1);
    }
    __syncthreads();

    const v8f vzero = {0.f, 0.f, 0.f, 0.f, 0.f, 0.f, 0.f, 0.f};
    v8f acc[4] = {vzero, vzero, vzero, vzero};
    const int mrow0 = wave * 16;
#pragma unroll
    for (int kb = 0; kb < 128; kb += 32) {
        v16bf afrag = load_a_bf16v<CS>(cin, mrow0, kb, lane);
#pragma unroll
        for (int t = 0; t < 4; ++t)
            acc[t] = wmma_bf16(afrag, load_b_frag(w1t, kb >> 5, t, lane), acc[t]);
    }
    __syncthreads();  // all waves done reading cin; safe to alias as h1

    const int c15 = lane & 15, hf = lane >> 4;
#pragma unroll
    for (int t = 0; t < 4; ++t) {
        float bb = fc1b[a * C_H + t * 16 + c15];
#pragma unroll
        for (int v = 0; v < 8; ++v) {
            float x = acc[t][v] + bb;
            x = x > 0.f ? x : 0.01f * x;            // leaky_relu(0.01)
            int row = mrow0 + hf * 8 + v;
            h1[row * C_H + t * 16 + c15] = x;
        }
    }
    __syncthreads();

    if (tid < 128) {
        int b = b0 + tid;
        const float* arow = actions + ((size_t)a * C_B + b) * C_A;
        int   idx  = 0;
        float best = arow[0];
#pragma unroll
        for (int k = 1; k < C_A; ++k) {
            float v = arow[k];
            if (v > best) { best = v; idx = k; }    // first-occurrence argmax
        }
        const float* w2 = fc2w + ((size_t)a * C_A + idx) * C_H;
        float q = fc2b[a * C_A + idx];
#pragma unroll
        for (int h = 0; h < C_H; ++h) q += h1[tid * C_H + h] * w2[h];
        out[(size_t)a * C_B + b] = q;
    }
}

// ---------------------------------------------------------------------------
extern "C" void kernel_launch(void* const* d_in, const int* in_sizes, int n_in,
                              void* d_out, int out_size, void* d_ws, size_t ws_size,
                              hipStream_t stream) {
    (void)in_sizes; (void)n_in; (void)out_size; (void)ws_size;
    const float* unary      = (const float*)d_in[0];
    const float* actions    = (const float*)d_in[1];
    const int*   edge_index = (const int*)d_in[2];
    const int*   edge_attr  = (const int*)d_in[3];
    const float* Wg         = (const float*)d_in[4];
    const float* rootg      = (const float*)d_in[5];
    const float* biasg      = (const float*)d_in[6];
    const float* fc1w       = (const float*)d_in[7];
    const float* fc1b       = (const float*)d_in[8];
    const float* fc2w       = (const float*)d_in[9];
    const float* fc2b       = (const float*)d_in[10];
    float*       out        = (float*)d_out;

    float* wts    = (float*)d_ws;        // 512 floats (padded to 1024)
    float* pooled = wts + 1024;          // NA*B*H = 262144 floats

    edge_weight_kernel<<<1, C_E0, 0, stream>>>(edge_index, edge_attr, wts);
    rgcn_pool_kernel<<<C_NA * C_B, 256, 0, stream>>>(unary, edge_index, edge_attr,
                                                     Wg, rootg, biasg, wts, pooled);
    mlp_kernel<<<dim3(C_B / 128, C_NA), 256, 0, stream>>>(actions, fc1w, fc1b,
                                                          fc2w, fc2b, pooled, out);
}
